// GeometricDriftField_56453050138729
// MI455X (gfx1250) — matrix operand order repaired and checked by reading, not compiled, not run on hardware
//
#include <hip/hip_runtime.h>

// ---------------- problem constants (from reference setup_inputs) -------------
#define BB      2
#define NN      4096
#define DD      2048
#define HH      16
#define DHH     128
#define KCONV   4
#define MROWS   (BB * NN)        // 8192
#define TD      (3 * DD)         // 6144
#define NCHUNK  16
#define CHUNKL  256

typedef __attribute__((ext_vector_type(16))) __bf16 v16bf;
typedef __attribute__((ext_vector_type(8)))  float  v8f;
typedef __attribute__((ext_vector_type(4)))  unsigned int u32x4;
typedef __attribute__((ext_vector_type(8)))  int i32x8;
typedef __attribute__((ext_vector_type(4)))  int i32x4;

union FragU { uint4 q[2]; v16bf v; };

__device__ __forceinline__ unsigned short f2bf(float f) {
    unsigned u = __float_as_uint(f);
    unsigned r = u + 0x7FFFu + ((u >> 16) & 1u);   // round-to-nearest-even
    return (unsigned short)(r >> 16);
}

__device__ __forceinline__ float softplus_f(float x) {
    return (x > 20.f) ? x : log1pf(expf(x));
}

// ---------------- K0: weight f32 -> bf16 --------------------------------------
__global__ __launch_bounds__(256) void cvt_w_kernel(const float* __restrict__ w,
                                                    unsigned short* __restrict__ o,
                                                    int n) {
    int i = blockIdx.x * 256 + threadIdx.x;
    if (i < n) o[i] = f2bf(w[i]);
}

// ---------------- K1: RMSNorm (one block per row) -----------------------------
__global__ __launch_bounds__(256) void rmsnorm_kernel(const float* __restrict__ x,
                                                      const float* __restrict__ scale,
                                                      float* __restrict__ xn) {
    const size_t row = blockIdx.x;
    const float* xr = x + row * (size_t)DD;
    float ss = 0.f;
    for (int i = threadIdx.x; i < DD; i += 256) { float v = xr[i]; ss += v * v; }
    for (int o = 16; o > 0; o >>= 1) ss += __shfl_down(ss, o, 32);
    __shared__ float red[8];
    __shared__ float tot;
    if ((threadIdx.x & 31) == 0) red[threadIdx.x >> 5] = ss;
    __syncthreads();
    if (threadIdx.x == 0) {
        float s = 0.f;
        for (int i = 0; i < 8; i++) s += red[i];
        tot = rsqrtf(s / (float)DD + 1e-6f);
    }
    __syncthreads();
    float r = tot;
    for (int i = threadIdx.x; i < DD; i += 256) {
        float v = scale[i] * xr[i] * r;
        v = fminf(fmaxf(v, -60000.f), 60000.f);
        xn[row * (size_t)DD + i] = v;
    }
}

// ---------------- K2: depthwise causal conv (K=4) + SiLU -> bf16 --------------
__global__ __launch_bounds__(256) void conv_silu_kernel(const float* __restrict__ xn,
                                                        const float* __restrict__ w,
                                                        const float* __restrict__ bias,
                                                        unsigned short* __restrict__ xc) {
    size_t i = (size_t)blockIdx.x * 256 + threadIdx.x;   // < MROWS*DD
    int d = (int)(i & (DD - 1));
    size_t nd = i >> 11;
    int n = (int)(nd & (NN - 1));
    int b = (int)(nd >> 12);
    float acc = bias[d];
#pragma unroll
    for (int k = 0; k < KCONV; k++) {
        int t = n - (KCONV - 1) + k;
        if (t >= 0)
            acc += xn[((size_t)b * NN + t) * DD + d] * w[d * KCONV + k];
    }
    float s = acc / (1.f + expf(-acc));                  // silu
    xc[i] = f2bf(s);
}

// ---------------- TDM descriptor: load 2D tile [128 rows x 32 bf16] to LDS ----
// Padded layout: pad_interval=3 (16 DWORDs = 64B) + pad_amount=3 (4 DWORDs = 16B)
// => LDS row stride 80B = 40 ushorts (bank-conflict-free, 16B aligned frags).
#if __has_builtin(__builtin_amdgcn_tensor_load_to_lds)
#define HAVE_TDM 1
__device__ __forceinline__ void tdm_load_tile(unsigned lds_addr, const void* gaddr,
                                              int ktot_elems) {
    unsigned long long ga = (unsigned long long)(size_t)gaddr;
    u32x4 g0;
    g0[0] = 1u;                                           // count=1, user mode
    g0[1] = lds_addr;                                     // LDS byte address
    g0[2] = (unsigned)(ga & 0xFFFFFFFFu);                 // global_addr[31:0]
    g0[3] = (unsigned)((ga >> 32) & 0x01FFFFFFu)          // global_addr[56:32]
          | 0x80000000u;                                  // type=2 ("image")
    i32x8 g1;
    g1[0] = (int)((1u << 16)     // data_size = 2 bytes
                | (1u << 20)     // pad_enable
                | (3u << 22)     // pad_interval: 16 DWORDs (64B) between pads
                | (3u << 25));   // pad_amount: 4 DWORDs (16B)
    g1[1] = (int)(32u << 16);    // tensor_dim0 = 32 elements (bits 79:48 lo)
    g1[2] = (int)(128u << 16);   // tensor_dim0 hi=0 | tensor_dim1 = 128
    g1[3] = (int)(32u << 16);    // tensor_dim1 hi=0 | tile_dim0 = 32
    g1[4] = 128;                 // tile_dim1 = 128, tile_dim2 = 0
    g1[5] = ktot_elems;          // tensor_dim0_stride[31:0] (elements)
    g1[6] = 0;                   // stride hi / tensor_dim1_stride lo
    g1[7] = 0;
    i32x4 z4 = {0, 0, 0, 0};
#if defined(__clang_major__) && (__clang_major__ >= 23)
    i32x8 z8 = {0, 0, 0, 0, 0, 0, 0, 0};
    __builtin_amdgcn_tensor_load_to_lds(g0, g1, z4, z4, z8, 0);
#else
    __builtin_amdgcn_tensor_load_to_lds(g0, g1, z4, z4, 0);
#endif
}
#else
#define HAVE_TDM 0
#endif

// ---------------- WMMA GEMM: C[M,Ntot] = A[M,K] * W[Ntot,K]^T (+residual) -----
// 128x128 block tile, 8 waves (2x4), each wave 64x32 = 8 WMMA 16x16 tiles.
// Double-buffered LDS; staging via Tensor Data Mover (TENSORcnt) when available.
template <bool RES>
__global__ __launch_bounds__(256) void gemm_bf16_kernel(const unsigned short* __restrict__ A,
                                                        const unsigned short* __restrict__ W,
                                                        const float* __restrict__ resid,
                                                        float* __restrict__ C,
                                                        int Ntot, int Ktot) {
    __shared__ __align__(128) unsigned short sA[2][128][40];   // 32 k + 16B TDM pad
    __shared__ __align__(128) unsigned short sB[2][128][40];
    const int t    = threadIdx.x;
    const int m0   = blockIdx.y * 128;
    const int n0   = blockIdx.x * 128;
    const int wv   = t >> 5;
    const int lane = t & 31;
    const int wm   = wv >> 2;     // 0..1
    const int wn   = wv & 3;      // 0..3
    const int rlo  = lane & 15;
    const int hi   = lane >> 4;

    v8f acc[4][2] = {};
    const int nsteps = Ktot >> 5;

#if HAVE_TDM
    // ---- prologue: DMA first K-slice into buffer 0 ----
    if (wv == 0) {
        tdm_load_tile((unsigned)(size_t)&sA[0][0][0], &A[(size_t)m0 * Ktot], Ktot);
        tdm_load_tile((unsigned)(size_t)&sB[0][0][0], &W[(size_t)n0 * Ktot], Ktot);
        __builtin_amdgcn_s_wait_tensorcnt(0);
    }
    __syncthreads();
#endif

    for (int s = 0; s < nsteps; s++) {
#if HAVE_TDM
        const int cur = s & 1;
        const int nxt = cur ^ 1;
        if (wv == 0 && (s + 1) < nsteps) {
            const int k0 = (s + 1) << 5;
            tdm_load_tile((unsigned)(size_t)&sA[nxt][0][0], &A[(size_t)m0 * Ktot + k0], Ktot);
            tdm_load_tile((unsigned)(size_t)&sB[nxt][0][0], &W[(size_t)n0 * Ktot + k0], Ktot);
        }
#else
        const int cur = 0;
        const int k0g = s << 5;
        // synchronous cooperative staging: 2 x 16B per thread per tile
#pragma unroll
        for (int i = 0; i < 2; i++) {
            int u  = t + i * 256;
            int r  = u >> 2;
            int co = (u & 3) * 8;
            *(uint4*)&sA[0][r][co] = *(const uint4*)&A[((size_t)(m0 + r)) * Ktot + k0g + co];
            *(uint4*)&sB[0][r][co] = *(const uint4*)&W[((size_t)(n0 + r)) * Ktot + k0g + co];
        }
        __syncthreads();
#endif
        FragU af[4], bf2[2];
#pragma unroll
        for (int mt = 0; mt < 4; mt++) {
            int row = wm * 64 + mt * 16 + rlo;
            af[mt].q[0] = *(const uint4*)&sA[cur][row][hi * 8];        // K 0..7   / 8..15
            af[mt].q[1] = *(const uint4*)&sA[cur][row][16 + hi * 8];   // K 16..23 / 24..31
        }
#pragma unroll
        for (int nt = 0; nt < 2; nt++) {
            int row = wn * 32 + nt * 16 + rlo;
            bf2[nt].q[0] = *(const uint4*)&sB[cur][row][hi * 8];       // K 0..7   / 8..15
            bf2[nt].q[1] = *(const uint4*)&sB[cur][row][16 + hi * 8];  // K 16..23 / 24..31
        }
#pragma unroll
        for (int mt = 0; mt < 4; mt++)
#pragma unroll
            for (int nt = 0; nt < 2; nt++)
                acc[mt][nt] = __builtin_amdgcn_wmma_f32_16x16x32_bf16(
                    false, af[mt].v, false, bf2[nt].v, (short)0, acc[mt][nt], false, false);

#if HAVE_TDM
        if (wv == 0) __builtin_amdgcn_s_wait_tensorcnt(0);  // next buffer landed
        __syncthreads();                                    // all waves done w/ cur + see nxt
#else
        __syncthreads();
#endif
    }

    // epilogue: C/D layout — VGPR i holds (M = i + 8*hi, N = lane&15)
    const int rbase = m0 + wm * 64 + hi * 8;
    const int cbase = n0 + wn * 32 + rlo;
#pragma unroll
    for (int mt = 0; mt < 4; mt++) {
#pragma unroll
        for (int nt = 0; nt < 2; nt++) {
#pragma unroll
            for (int i = 0; i < 8; i++) {
                int row = rbase + mt * 16 + i;
                int col = cbase + nt * 16;
                float v = acc[mt][nt][i];
                if (RES) v += resid[(size_t)row * Ntot + col];
                C[(size_t)row * Ntot + col] = v;
            }
        }
    }
}

// ---------------- K3: dt=clip(softplus(+bias)), RoPE, v_in = rot(v)*dt --------
__global__ __launch_bounds__(256) void rope_dt_kernel(float* __restrict__ fused,
                                                      const float* __restrict__ dt_bias,
                                                      float* __restrict__ v_in) {
    size_t i = (size_t)blockIdx.x * 256 + threadIdx.x;   // < MROWS * H * 64
    int j = (int)(i & 63);
    size_t r = i >> 6;
    int h = (int)(r & (HH - 1));
    size_t mrow = r >> 4;                // b*NN + n
    int n = (int)(mrow & (NN - 1));

    float inv = powf(10000.f, -((float)(2 * j) / (float)DHH));
    float ang = (float)n * inv;
    float cs = cosf(ang), sn = sinf(ang);

    int d1 = h * DHH + j;
    int d2 = d1 + DHH / 2;
    float* frow = fused + mrow * (size_t)TD;

    float dt1 = fminf(fmaxf(softplus_f(frow[d1] + dt_bias[d1]), 0.001f), 2.0f);
    float dt2 = fminf(fmaxf(softplus_f(frow[d2] + dt_bias[d2]), 0.001f), 2.0f);
    frow[d1] = dt1;                      // overwrite dt_raw region with dt
    frow[d2] = dt2;

    float v1 = frow[DD + d1];
    float v2 = frow[DD + d2];
    float o1 = v1 * cs - v2 * sn;
    float o2 = v1 * sn + v2 * cs;
    v_in[mrow * (size_t)DD + d1] = o1 * dt1;
    v_in[mrow * (size_t)DD + d2] = o2 * dt2;
}

// ---------------- K4: per-(b,chunk,d) 256-step scan ---------------------------
__global__ __launch_bounds__(256) void chunk_scan_kernel(const float* __restrict__ fused,
                                                         float* __restrict__ vio,
                                                         float* __restrict__ expL,
                                                         float* __restrict__ bd,
                                                         float* __restrict__ bo) {
    size_t i = (size_t)blockIdx.x * 256 + threadIdx.x;   // < BB*NCHUNK*DD = 65536
    int d = (int)(i & (DD - 1));
    size_t r = i >> 11;
    int c = (int)(r & (NCHUNK - 1));
    int b = (int)(r >> 4);
    float L = 0.f, hs = 0.f;
    size_t base = (size_t)b * NN + (size_t)c * CHUNKL;
    for (int t = 0; t < CHUNKL; t++) {
        size_t row = base + t;
        float la = -fused[row * (size_t)TD + d];
        float Ln = fmaxf(L + la, -20.f);                 // running clip == clip(cumsum)
        hs = expf(Ln - L) * hs + vio[row * (size_t)DD + d];
        vio[row * (size_t)DD + d] = hs;
        expL[row * (size_t)DD + d] = expf(Ln);
        L = Ln;
    }
    bd[i] = L;
    bo[i] = hs;
}

// ---------------- K5: cross-chunk carry combine per (b,d) ---------------------
__global__ __launch_bounds__(256) void combine_kernel(const float* __restrict__ bd,
                                                      const float* __restrict__ bo,
                                                      float* __restrict__ carries) {
    int i = blockIdx.x * 256 + threadIdx.x;              // < BB*DD = 4096
    int d = i & (DD - 1);
    int b = i >> 11;
    float cd[NCHUNK];
    float cum = 0.f, stab = -1e30f;
    for (int c = 0; c < NCHUNK; c++) {
        cum += bd[((size_t)(b * NCHUNK + c)) * DD + d];
        float v = fminf(fmaxf(cum, -80.f), 0.f);
        cd[c] = v;
        stab = fmaxf(stab, v);
    }
    float run = 0.f;
    for (int c = 0; c < NCHUNK; c++) {
        float ncd = fminf(fmaxf(cd[c] - stab, -20.f), 0.f);
        carries[((size_t)(b * NCHUNK + c)) * DD + d] = run * expf(ncd);
        run += bo[((size_t)(b * NCHUNK + c)) * DD + d] * expf(-ncd);
    }
}

// ---------------- K6: apply carries, head-mix, sigmoid gate -> bf16 -----------
__global__ __launch_bounds__(256) void mix_gate_kernel(const float* __restrict__ chunk_out,
                                                       const float* __restrict__ expL,
                                                       const float* __restrict__ carries,
                                                       const float* __restrict__ fused,
                                                       const float* __restrict__ head_mix,
                                                       unsigned short* __restrict__ a2) {
    size_t i = (size_t)blockIdx.x * 256 + threadIdx.x;   // < MROWS*DD
    int dd = (int)(i & (DHH - 1));
    size_t r = i >> 7;
    int m = (int)(r & (HH - 1));
    size_t row = r >> 4;                                  // b*NN + n
    int n = (int)(row & (NN - 1));
    int b = (int)(row >> 12);
    int c = n >> 8;
    float vacc = 0.f;
#pragma unroll
    for (int h = 0; h < HH; h++) {
        int d = h * DHH + dd;
        size_t idx = row * (size_t)DD + d;
        float fin = chunk_out[idx] + carries[((size_t)(b * NCHUNK + c)) * DD + d] * expL[idx];
        vacc += fin * head_mix[h * HH + m];
    }
    int dout = m * DHH + dd;
    float g = fused[row * (size_t)TD + 2 * DD + dout];
    g = 1.f / (1.f + expf(-g));
    a2[row * (size_t)DD + dout] = f2bf(vacc * g);
}

// ---------------- launch ------------------------------------------------------
extern "C" void kernel_launch(void* const* d_in, const int* in_sizes, int n_in,
                              void* d_out, int out_size, void* d_ws, size_t ws_size,
                              hipStream_t stream) {
    const float* x          = (const float*)d_in[0];
    const float* norm_scale = (const float*)d_in[1];
    const float* conv_w     = (const float*)d_in[2];
    const float* conv_b     = (const float*)d_in[3];
    const float* in_proj_w  = (const float*)d_in[4];
    const float* dt_bias    = (const float*)d_in[5];
    const float* head_mix   = (const float*)d_in[6];
    const float* out_proj_w = (const float*)d_in[7];
    float* out = (float*)d_out;

    char* ws = (char*)d_ws;
    const size_t MiB = 1024ull * 1024ull;
    // buffer carve (with reuse): total ~386 MiB
    float*          xnF   = (float*)(ws + 0);                       // 64 MiB (reused as expL)
    float*          expL  = (float*)(ws + 0);                       // alias: xn dead after conv
    unsigned short* xcBf  = (unsigned short*)(ws + 64 * MiB);       // 32 MiB (reused as a2)
    unsigned short* a2Bf  = (unsigned short*)(ws + 64 * MiB);       // alias: xc dead after GEMM1
    float*          fused = (float*)(ws + 96 * MiB);                // 192 MiB
    float*          vio   = (float*)(ws + 288 * MiB);               // 64 MiB (v_in -> chunk_out)
    unsigned short* wp1   = (unsigned short*)(ws + 352 * MiB);      // 24 MiB
    unsigned short* wp2   = (unsigned short*)(ws + 376 * MiB);      // 8 MiB
    float*          bd    = (float*)(ws + 384 * MiB);               // 256 KiB
    float*          bo    = (float*)(ws + 384 * MiB + 256 * 1024);  // 256 KiB
    float*          carr  = (float*)(ws + 384 * MiB + 512 * 1024);  // 256 KiB

    // K0: weights -> bf16
    cvt_w_kernel<<<(TD * DD + 255) / 256, 256, 0, stream>>>(in_proj_w, wp1, TD * DD);
    cvt_w_kernel<<<(DD * DD + 255) / 256, 256, 0, stream>>>(out_proj_w, wp2, DD * DD);

    // K1: RMSNorm
    rmsnorm_kernel<<<MROWS, 256, 0, stream>>>(x, norm_scale, xnF);

    // K2: conv + silu -> bf16 A operand
    conv_silu_kernel<<<(MROWS * DD) / 256, 256, 0, stream>>>(xnF, conv_w, conv_b, xcBf);

    // GEMM1: fused = xc @ in_proj_w^T   (8192 x 2048 x 6144)
    gemm_bf16_kernel<false><<<dim3(TD / 128, MROWS / 128), 256, 0, stream>>>(
        xcBf, wp1, nullptr, fused, TD, DD);

    // K3: dt/softplus (in place), RoPE, v_in
    rope_dt_kernel<<<(MROWS * HH * 64) / 256, 256, 0, stream>>>(fused, dt_bias, vio);

    // K4: intra-chunk scan
    chunk_scan_kernel<<<(BB * NCHUNK * DD) / 256, 256, 0, stream>>>(fused, vio, expL, bd, bo);

    // K5: cross-chunk combine
    combine_kernel<<<(BB * DD) / 256, 256, 0, stream>>>(bd, bo, carr);

    // K6: apply carries + head mix + gate -> bf16 A operand
    mix_gate_kernel<<<(MROWS * DD) / 256, 256, 0, stream>>>(vio, expL, carr, fused, head_mix, a2Bf);

    // GEMM2 + residual: out = residual + (a2 @ out_proj_w^T)   (8192 x 2048 x 2048)
    gemm_bf16_kernel<true><<<dim3(DD / 128, MROWS / 128), 256, 0, stream>>>(
        a2Bf, wp2, x, out, DD, DD);
}
